// AutoEncoder_36696200577712
// MI455X (gfx1250) — matrix operand order repaired and checked by reading, not compiled, not run on hardware
//
#include <hip/hip_runtime.h>
#include <hip/hip_bf16.h>
#include <stdint.h>

// ---------------------------------------------------------------------------
// Routed (MoE) autoencoder: per-expert fused 1024->512->128->512->1024 MLP.
// bf16 WMMA (v_wmma_f32_16x16x32_bf16), fp32 accumulate, fp32 out.
// ---------------------------------------------------------------------------

#define N_ATOMS 32768
#define D_IN    1024
#define H1D     512
#define LATD    128
#define D_OUT   1024
#define NEXP    4

typedef __bf16 bf16_t;
typedef __attribute__((ext_vector_type(16))) __bf16 bf16x16;
typedef __attribute__((ext_vector_type(8)))  float  floatx8;

union Frag { bf16x16 v; uint4 q[2]; };

__device__ __forceinline__ uint32_t f32_to_bf16_bits(float f) {
    uint32_t u = __float_as_uint(f);
    u += 0x7FFFu + ((u >> 16) & 1u);   // round-to-nearest-even
    return u >> 16;
}

// ------------------------------ routing ------------------------------------
__global__ void k_zero(int* p) {
    if (threadIdx.x < 12) p[threadIdx.x] = 0;
}

__global__ void k_count(const int* __restrict__ sym, int* __restrict__ counts, int n) {
    int i = blockIdx.x * blockDim.x + threadIdx.x;
    if (i < n) atomicAdd(&counts[sym[i]], 1);
}

__global__ void k_offsets(const int* __restrict__ counts, int* __restrict__ offsets) {
    if (threadIdx.x == 0) {
        int acc = 0;
        for (int e = 0; e < NEXP; ++e) { offsets[e] = acc; acc += counts[e]; }
    }
}

__global__ void k_scatter(const int* __restrict__ sym, int* __restrict__ cursors,
                          const int* __restrict__ offsets, int* __restrict__ perm, int n) {
    int i = blockIdx.x * blockDim.x + threadIdx.x;
    if (i < n) {
        int e = sym[i];
        int pos = atomicAdd(&cursors[e], 1);
        perm[offsets[e] + pos] = i;
    }
}

// --------------------------- weight repacking ------------------------------
// Pack fp32 W[e][K][Nn] into bf16 WMMA B-fragments, layout [e][nt][kt][lane][j]
// (dwords). Element (k,n) of a 32x16 tile lives at lane = (n%16) + 16*(k/16),
// dword j = (k%16)/2, half = k%2  -> two global_load_b128 per fragment.
__global__ void k_pack(const float* __restrict__ W, uint32_t* __restrict__ dst,
                       int K, int Nn) {
    const int KT = K >> 5, NT = Nn >> 4;
    const long total = (long)NEXP * NT * KT * 256;
    long t = (long)blockIdx.x * blockDim.x + threadIdx.x;
    if (t >= total) return;
    int j    = (int)(t & 7);
    int lane = (int)((t >> 3) & 31);
    long rest = t >> 8;
    int kt = (int)(rest % KT); rest /= KT;
    int nt = (int)(rest % NT);
    int e  = (int)(rest / NT);
    int k = kt * 32 + 2 * j + ((lane >> 4) << 4);
    int n = nt * 16 + (lane & 15);
    const float* base = W + (size_t)e * K * Nn;
    uint32_t lo = f32_to_bf16_bits(base[(size_t)k * Nn + n]);
    uint32_t hi = f32_to_bf16_bits(base[(size_t)(k + 1) * Nn + n]);
    dst[t] = lo | (hi << 16);
}

// ------------------------------ main kernel --------------------------------
// LDS pitches (bf16 elems), padded so pitch%64dw == 4 -> conflict-free b128.
#define PITCH_A 1032   // X / decoder-hidden region (width 1024 / 512)
#define PITCH_H 520    // encoder hidden (width 512)
#define PITCH_Z 136    // latent (width 128)
#define MT 64          // rows per block
#define SMEM_BYTES (256 + MT*PITCH_A*2 + MT*PITCH_H*2 + MT*PITCH_Z*2) // 216320

__device__ __forceinline__ floatx8 gemm_tile(const bf16_t* __restrict__ Abase, int apitch,
                                             const uint32_t* __restrict__ Bexp,
                                             int nt, int KT, int strip, int lane) {
    floatx8 acc = {};
    const int row = strip * 16 + (lane & 15);
    const bf16_t* arow = Abase + (size_t)row * apitch + 16 * (lane >> 4);
    const uint32_t* bp = Bexp + (size_t)nt * KT * 256 + lane * 8;
    #pragma unroll 4
    for (int kt = 0; kt < KT; ++kt) {
        Frag a, b;
        a.q[0] = *(const uint4*)(arow + kt * 32);        // K 0..7   (16..23 hi lanes)
        a.q[1] = *(const uint4*)(arow + kt * 32 + 8);    // K 8..15  (24..31 hi lanes)
        b.q[0] = *(const uint4*)(bp);
        b.q[1] = *(const uint4*)(bp + 4);
        bp += 256;
        acc = __builtin_amdgcn_wmma_f32_16x16x32_bf16(
                  false, a.v, false, b.v, (short)0, acc, false, false);
    }
    return acc;
}

__device__ __forceinline__ void epilogue_lds(bf16_t* __restrict__ dst, int pitch,
                                             int strip, int lane, int nt, floatx8 acc,
                                             const float* __restrict__ bias) {
    const int col = nt * 16 + (lane & 15);
    const float b = bias[col];
    const int rbase = strip * 16 + 8 * (lane >> 4);
    #pragma unroll
    for (int r = 0; r < 8; ++r) {
        float v = acc[r] + b;
        v = fmaxf(v, 0.0f);                          // ReLU
        dst[(size_t)(rbase + r) * pitch + col] = (bf16_t)v;
    }
}

__global__ void __launch_bounds__(256)
k_moe(const float* __restrict__ X,
      const int* __restrict__ perm, const int* __restrict__ counts,
      const int* __restrict__ offsets,
      const uint32_t* __restrict__ pw1, const uint32_t* __restrict__ pw2,
      const uint32_t* __restrict__ pw3, const uint32_t* __restrict__ pw4,
      const float* __restrict__ be1, const float* __restrict__ be2,
      const float* __restrict__ bd1, const float* __restrict__ bd2,
      float* __restrict__ out) {
    extern __shared__ char smem[];
    int*    rowIdxS = (int*)smem;                                  // 64 ints
    bf16_t* As = (bf16_t*)(smem + 256);                            // 64 x 1032
    bf16_t* Hs = (bf16_t*)(smem + 256 + MT * PITCH_A * 2);         // 64 x 520
    bf16_t* Zs = (bf16_t*)(smem + 256 + MT * (PITCH_A + PITCH_H) * 2); // 64 x 136
    bf16_t* Ds = As;  // decoder hidden reuses X region (X dead after layer 1)

    const int e   = blockIdx.y;
    const int cnt = counts[e];
    const int base = blockIdx.x * MT;
    if (base >= cnt) return;
    const int valid = min(MT, cnt - base);
    const int off = offsets[e];

    const int lane  = threadIdx.x & 31;
    const int wave  = threadIdx.x >> 5;
    const int strip = wave & 3;     // 4 strips of 16 rows
    const int half  = wave >> 2;    // 2-way split of the N dimension

    if (threadIdx.x < MT)
        rowIdxS[threadIdx.x] = perm[off + base + min((int)threadIdx.x, valid - 1)];
    __syncthreads();

    // ---- stage X tile -> LDS bf16 (coalesced float4 gather) ----
    const float4* X4 = (const float4*)X;
    for (int f = threadIdx.x; f < MT * (D_IN / 4); f += blockDim.x) {
        int row = f >> 8, c4 = f & 255;
        float4 v = X4[(size_t)rowIdxS[row] * (D_IN / 4) + c4];
        uint32_t d0 = f32_to_bf16_bits(v.x) | (f32_to_bf16_bits(v.y) << 16);
        uint32_t d1 = f32_to_bf16_bits(v.z) | (f32_to_bf16_bits(v.w) << 16);
        *(uint2*)(As + (size_t)row * PITCH_A + c4 * 4) = make_uint2(d0, d1);
    }
    __syncthreads();

    const uint32_t* B1e = pw1 + (size_t)e * (H1D / 16)   * (D_IN / 32) * 256;
    const uint32_t* B2e = pw2 + (size_t)e * (LATD / 16)  * (H1D / 32)  * 256;
    const uint32_t* B3e = pw3 + (size_t)e * (H1D / 16)   * (LATD / 32) * 256;
    const uint32_t* B4e = pw4 + (size_t)e * (D_OUT / 16) * (H1D / 32)  * 256;

    // ---- layer 1: X[64x1024] x W1[1024x512] -> H (ReLU) ----
    for (int i = 0; i < 16; ++i) {
        int nt = half * 16 + i;
        floatx8 acc = gemm_tile(As, PITCH_A, B1e, nt, D_IN / 32, strip, lane);
        epilogue_lds(Hs, PITCH_H, strip, lane, nt, acc, be1 + (size_t)e * H1D);
    }
    __syncthreads();

    // ---- layer 2: H[64x512] x W2[512x128] -> Z (ReLU) ----
    for (int i = 0; i < 4; ++i) {
        int nt = half * 4 + i;
        floatx8 acc = gemm_tile(Hs, PITCH_H, B2e, nt, H1D / 32, strip, lane);
        epilogue_lds(Zs, PITCH_Z, strip, lane, nt, acc, be2 + (size_t)e * LATD);
    }
    __syncthreads();

    // ---- layer 3: Z[64x128] x W3[128x512] -> D (ReLU), overwrites X region ----
    for (int i = 0; i < 16; ++i) {
        int nt = half * 16 + i;
        floatx8 acc = gemm_tile(Zs, PITCH_Z, B3e, nt, LATD / 32, strip, lane);
        epilogue_lds(Ds, PITCH_H, strip, lane, nt, acc, bd1 + (size_t)e * H1D);
    }
    __syncthreads();

    // ---- layer 4: D[64x512] x W4[512x1024] -> out (bias, no ReLU, scatter) ----
    for (int i = 0; i < 32; ++i) {
        int nt = half * 32 + i;
        floatx8 acc = gemm_tile(Ds, PITCH_H, B4e, nt, H1D / 32, strip, lane);
        const int col = nt * 16 + (lane & 15);
        const float b = bd2[(size_t)e * D_OUT + col];
        const int rbase = strip * 16 + 8 * (lane >> 4);
        #pragma unroll
        for (int r = 0; r < 8; ++r) {
            int lr = rbase + r;
            if (lr < valid)
                out[(size_t)rowIdxS[lr] * D_OUT + col] = acc[r] + b;
        }
    }
}

// ------------------------------ launch -------------------------------------
extern "C" void kernel_launch(void* const* d_in, const int* in_sizes, int n_in,
                              void* d_out, int out_size, void* d_ws, size_t ws_size,
                              hipStream_t stream) {
    (void)in_sizes; (void)n_in; (void)out_size; (void)ws_size;
    const float* X    = (const float*)d_in[0];
    const int*   sym  = (const int*)d_in[1];
    const float* We1  = (const float*)d_in[2];
    const float* be1  = (const float*)d_in[3];
    const float* We2  = (const float*)d_in[4];
    const float* be2  = (const float*)d_in[5];
    const float* Wd1  = (const float*)d_in[6];
    const float* bd1  = (const float*)d_in[7];
    const float* Wd2  = (const float*)d_in[8];
    const float* bd2  = (const float*)d_in[9];
    float* out = (float*)d_out;

    // workspace layout
    int* counts  = (int*)d_ws;        // [4]
    int* cursors = counts + 4;        // [4]
    int* offsets = counts + 8;        // [4]
    int* perm    = (int*)((char*)d_ws + 256);                       // [N]
    uint32_t* pw1 = (uint32_t*)((char*)d_ws + 256 + N_ATOMS * 4);   // 4 MiB
    uint32_t* pw2 = pw1 + (size_t)NEXP * (H1D/16)  * (D_IN/32) * 256;
    uint32_t* pw3 = pw2 + (size_t)NEXP * (LATD/16) * (H1D/32)  * 256;
    uint32_t* pw4 = pw3 + (size_t)NEXP * (H1D/16)  * (LATD/32) * 256;

    // routing
    k_zero<<<1, 32, 0, stream>>>(counts);
    k_count<<<N_ATOMS / 256, 256, 0, stream>>>(sym, counts, N_ATOMS);
    k_offsets<<<1, 1, 0, stream>>>(counts, offsets);
    k_scatter<<<N_ATOMS / 256, 256, 0, stream>>>(sym, cursors, offsets, perm, N_ATOMS);

    // weight repack (fp32 -> bf16 WMMA fragment layout)
    k_pack<<<(NEXP * (D_IN/32)  * (H1D/16)   * 256) / 256, 256, 0, stream>>>(We1, pw1, D_IN, H1D);
    k_pack<<<(NEXP * (H1D/32)   * (LATD/16)  * 256) / 256, 256, 0, stream>>>(We2, pw2, H1D, LATD);
    k_pack<<<(NEXP * (LATD/32)  * (H1D/16)   * 256) / 256, 256, 0, stream>>>(Wd1, pw3, LATD, H1D);
    k_pack<<<(NEXP * (H1D/32)   * (D_OUT/16) * 256) / 256, 256, 0, stream>>>(Wd2, pw4, H1D, D_OUT);

    // fused per-expert MLP
    dim3 grid((N_ATOMS + MT - 1) / MT, NEXP);
    k_moe<<<grid, 256, SMEM_BYTES, stream>>>(X, perm, counts, offsets,
                                             pw1, pw2, pw3, pw4,
                                             be1, be2, bd1, bd2, out);
}
